// PerceiverAttention_11261404250349
// MI455X (gfx1250) — matrix operand order repaired and checked by reading, not compiled, not run on hardware
//
#include <hip/hip_runtime.h>
#include <hip/hip_bf16.h>

typedef __bf16 bf16;
typedef __attribute__((ext_vector_type(16))) __bf16 v16bf;
typedef __attribute__((ext_vector_type(8)))  __bf16 v8bf;
typedef __attribute__((ext_vector_type(8)))  float  v8f;

#define B_      8
#define M_      4096
#define NLAT    64
#define DIM_    1024
#define DH_     128
#define H_      8
#define INNER_  1024
#define S_      (M_ + NLAT)     /* 4160 */
#define KV2_    2048
#define SPLIT_  10
#define SCHUNK_ (S_ / SPLIT_)   /* 416 = 13 * 32 */
#define EPS_    1e-5f

static __device__ __forceinline__ v16bf cat8(v8bf lo, v8bf hi) {
    v16bf r;
#pragma unroll
    for (int i = 0; i < 8; i++) { r[i] = lo[i]; r[i + 8] = hi[i]; }
    return r;
}

static __device__ __forceinline__ v8f zero8() {
    v8f z;
#pragma unroll
    for (int i = 0; i < 8; i++) z[i] = 0.0f;
    return z;
}

static __device__ __forceinline__ v8f wmma_bf(v16bf a, v16bf b, v8f c) {
    return __builtin_amdgcn_wmma_f32_16x16x32_bf16(false, a, false, b, (short)0, c, false, false);
}

// contiguous 32B (32B-aligned) B-fragment load: two global_load_b128 straight
// into the adjacent VGPR pair WMMA consumes (no assembly moves).
static __device__ __forceinline__ v16bf ld16(const bf16* p) {
    return *(const v16bf*)p;
}

// ---------------------------------------------------------------------------
// LayerNorm of x and latents, written bf16 into the fused KV-input buffer:
// kv_in[b, 0..M) = LN(x[b]),  kv_in[b, M..S) = LN(latents[b]).
// One 256-thread block per row of 1024 elements.
// ---------------------------------------------------------------------------
__global__ __launch_bounds__(256) void ln_fuse_kernel(
    const float* __restrict__ x, const float* __restrict__ lat,
    const float* __restrict__ gm, const float* __restrict__ bm,
    const float* __restrict__ gl, const float* __restrict__ bl,
    bf16* __restrict__ kv_in)
{
    int row = blockIdx.x;          // 0 .. B*S-1
    int b = row / S_;
    int i = row % S_;
    const float* src; const float* g; const float* be;
    if (i < M_) { src = x   + ((size_t)b * M_   +  i       ) * DIM_; g = gm; be = bm; }
    else        { src = lat + ((size_t)b * NLAT + (i - M_) ) * DIM_; g = gl; be = bl; }

    int tid = threadIdx.x;
    float v[4]; float s = 0.f, s2 = 0.f;
#pragma unroll
    for (int j = 0; j < 4; j++) {
        int c = j * 256 + tid;
        float t = src[c]; v[j] = t; s += t; s2 += t * t;
    }
#pragma unroll
    for (int off = 16; off > 0; off >>= 1) {
        s  += __shfl_xor(s,  off, 32);
        s2 += __shfl_xor(s2, off, 32);
    }
    __shared__ float red[16];
    int wid = tid >> 5;
    if ((tid & 31) == 0) { red[wid] = s; red[8 + wid] = s2; }
    __syncthreads();
    float ts = 0.f, t2 = 0.f;
#pragma unroll
    for (int w = 0; w < 8; w++) { ts += red[w]; t2 += red[8 + w]; }
    float mu  = ts * (1.0f / DIM_);
    float var = t2 * (1.0f / DIM_) - mu * mu;
    float rs  = rsqrtf(var + EPS_);
    bf16* dst = kv_in + (size_t)row * DIM_;
#pragma unroll
    for (int j = 0; j < 4; j++) {
        int c = j * 256 + tid;
        dst[c] = (bf16)((v[j] - mu) * rs * g[c] + be[c]);
    }
}

// ---------------------------------------------------------------------------
// Tiled f32 -> bf16 transposing convert: in [K][N] f32  ->  outT [N][K] bf16.
// Weights land transposed so WMMA B-fragments (lane = N col, elems = K) are
// contiguous 32B global reads; W stays resident in the 192MB L2.
// ---------------------------------------------------------------------------
__global__ __launch_bounds__(256) void cvt_transpose_kernel(
    const float* __restrict__ in, bf16* __restrict__ outT, int K, int N)
{
    __shared__ float tile[32][33];
    int k0 = blockIdx.y * 32, n0 = blockIdx.x * 32;
    int tx = threadIdx.x & 31, ty = threadIdx.x >> 5;   // 32 x 8
#pragma unroll
    for (int j = 0; j < 32; j += 8)
        tile[ty + j][tx] = in[(size_t)(k0 + ty + j) * N + n0 + tx];
    __syncthreads();
#pragma unroll
    for (int j = 0; j < 32; j += 8)
        outT[(size_t)(n0 + ty + j) * K + k0 + tx] = (bf16)tile[tx][ty + j];
}

// ---------------------------------------------------------------------------
// LDS-free, barrier-free WMMA bf16 GEMM.
//   C[row, n] = (sum_k A[phys(row), k] * Wt[n, k] + bias[n]) * scale
// phys(row) = (row/rpb)*bstride + roff + (row%rpb)  (q-projection sub-view).
// Block tile 128x128, 8 waves (2x4), wave tile 64x32 (4x2 of 16x16).
// A-frags: lane = M row, two adjacent 16B chunks (contiguous 32B) per K=32 step.
// B-frags: lane = N col, one contiguous 32B chunk per K=32 step (Wt transposed).
// MODE 0: bf16 row-major out. MODE 1: f32 row-major out.
// MODE 2: split KV — n<1024 -> k_bf[row][n] (row-major),
//                    n>=1024 -> vt[(b*H+h)*128+d][s] (transposed, packed 16B).
// ---------------------------------------------------------------------------
template <int MODE>
__global__ __launch_bounds__(256) void gemm_wmma_kernel(
    const bf16* __restrict__ A, const bf16* __restrict__ Wt,
    const float* __restrict__ bias, void* __restrict__ Cout, void* __restrict__ Cout2,
    int K, int Ncols, int rpb, int bstride, int roff, float scale)
{
    int tid  = threadIdx.x;
    int lane = tid & 31, lm = lane & 15, kh = lane >> 4;
    int wid  = tid >> 5;
    int wm   = wid >> 2;           // 0..1  (64 rows each)
    int wn   = wid & 3;            // 0..3  (32 cols each)
    int rowBase = blockIdx.y * 128;
    int nBase   = blockIdx.x * 128;

    v8f acc[4][2];
#pragma unroll
    for (int mt = 0; mt < 4; mt++)
#pragma unroll
        for (int nt = 0; nt < 2; nt++) acc[mt][nt] = zero8();

    const bf16* arow[4];
#pragma unroll
    for (int mt = 0; mt < 4; mt++) {
        int lrow = rowBase + wm * 64 + mt * 16 + lm;
        size_t prow = (size_t)(lrow / rpb) * bstride + roff + (lrow % rpb);
        arow[mt] = A + prow * (size_t)K;
    }
    const bf16* bcol[2];
#pragma unroll
    for (int nt = 0; nt < 2; nt++)
        bcol[nt] = Wt + (size_t)(nBase + wn * 32 + nt * 16 + lm) * K;

    for (int k0 = 0; k0 < K; k0 += 32) {
        v16bf af[4], bfm[2];
#pragma unroll
        for (int mt = 0; mt < 4; mt++) {
            const bf16* p = arow[mt] + k0 + kh * 16;
            af[mt] = cat8(*(const v8bf*)p, *(const v8bf*)(p + 8));
        }
#pragma unroll
        for (int nt = 0; nt < 2; nt++)
            bfm[nt] = ld16(bcol[nt] + k0 + kh * 16);
        if (k0 + 32 < K) {
            __builtin_prefetch(arow[0] + k0 + 32, 0, 1);   // global_prefetch_b8
            __builtin_prefetch(bcol[0] + k0 + 32, 0, 1);
        }
#pragma unroll
        for (int mt = 0; mt < 4; mt++)
#pragma unroll
            for (int nt = 0; nt < 2; nt++)
                acc[mt][nt] = wmma_bf(af[mt], bfm[nt], acc[mt][nt]);
    }

    // epilogue: C layout = VGPR r -> M = r + 8*kh, N = lane%16
#pragma unroll
    for (int mt = 0; mt < 4; mt++) {
#pragma unroll
        for (int nt = 0; nt < 2; nt++) {
            int n = nBase + wn * 32 + nt * 16 + lm;
            float bval = bias[n];
            if (MODE == 2 && nBase >= INNER_) {
                // V half: packed transposed store (8 consecutive s per lane)
                int sbase0 = rowBase + wm * 64 + mt * 16 + 8 * kh;
                int b = sbase0 / S_;
                int s = sbase0 % S_;           // 8-run never straddles (S_%8==0)
                int hd = n - INNER_;           // (h*128 + d)
                union { uint4 u; bf16 h8[8]; } pk;
#pragma unroll
                for (int r = 0; r < 8; r++)
                    pk.h8[r] = (bf16)((acc[mt][nt][r] + bval) * scale);
                bf16* vt = (bf16*)Cout2;
                *(uint4*)(vt + (size_t)(b * (H_ * DH_) + hd) * S_ + s) = pk.u;
            } else {
#pragma unroll
                for (int r = 0; r < 8; r++) {
                    int m = wm * 64 + mt * 16 + r + 8 * kh;
                    size_t row = (size_t)rowBase + m;
                    float vv = (acc[mt][nt][r] + bval) * scale;
                    if (MODE == 1)      ((float*)Cout)[row * Ncols + n] = vv;
                    else if (MODE == 2) ((bf16*)Cout)[row * (size_t)INNER_ + n] = (bf16)vv;
                    else                ((bf16*)Cout)[row * Ncols + n] = (bf16)vv;
                }
            }
        }
    }
}

// ---------------------------------------------------------------------------
// Flash-attention partials. Grid (SPLIT, H, B), 128 threads = 4 independent
// waves (no block barriers). Wave mt owns q rows [mt*16, mt*16+16).
// q pre-scaled by DH^-0.5. k read row-major (sim B-frags contiguous since
// K-dim == head dim); V read from the pre-transposed vt buffer (P@V B-frags
// contiguous). Only LDS use: per-wave P relayout C->A, ordered by the
// architectural in-order DS guarantee + s_wait_dscnt.
// ---------------------------------------------------------------------------
__global__ __launch_bounds__(128) void attn_partial_kernel(
    const bf16* __restrict__ q, const bf16* __restrict__ kmat,
    const bf16* __restrict__ vt,
    float* __restrict__ pout, float* __restrict__ pm, float* __restrict__ pl)
{
    __shared__ bf16 Pld[4][16][40];    // per-wave P tile [m][s]

    int tid  = threadIdx.x;
    int lane = tid & 31, lm = lane & 15, kh = lane >> 4;
    int mt   = tid >> 5;               // wave id = q row tile
    int sp = blockIdx.x, h = blockIdx.y, b = blockIdx.z;
    int bh = b * H_ + h;

    const bf16* qbase = q    + ((size_t)b * NLAT) * INNER_ + h * DH_;
    const bf16* kbase = kmat + (size_t)b * S_ * INNER_ + h * DH_;
    const bf16* vtb   = vt   + (size_t)bh * DH_ * S_;

    // preload q A-fragments (4 chunks of K=32 over DH=128)
    v16bf afq[4];
#pragma unroll
    for (int dc = 0; dc < 4; dc++) {
        const bf16* p = qbase + (size_t)(mt * 16 + lm) * INNER_ + dc * 32 + kh * 16;
        afq[dc] = cat8(*(const v8bf*)p, *(const v8bf*)(p + 8));
    }

    v8f oacc[8];
#pragma unroll
    for (int n8 = 0; n8 < 8; n8++) oacc[n8] = zero8();
    float mi[8], li[8];
#pragma unroll
    for (int r = 0; r < 8; r++) { mi[r] = -3.0e38f; li[r] = 0.f; }

    int s0 = sp * SCHUNK_;
    for (int t = 0; t < SCHUNK_ / 32; t++, s0 += 32) {
        // sim = q @ k^T : 16 x 32 tile per wave
        v8f sim[2] = { zero8(), zero8() };
#pragma unroll
        for (int dc = 0; dc < 4; dc++) {
#pragma unroll
            for (int nt = 0; nt < 2; nt++) {
                const bf16* kp = kbase + (size_t)(s0 + nt * 16 + lm) * INNER_ + dc * 32 + kh * 16;
                sim[nt] = wmma_bf(afq[dc], ld16(kp), sim[nt]);
            }
        }
        if (t + 1 < SCHUNK_ / 32) {
            __builtin_prefetch(kbase + (size_t)(s0 + 32 + lm) * INNER_, 0, 1);
            __builtin_prefetch(vtb + (size_t)lm * S_ + s0 + 32, 0, 1);
        }

        // online softmax per row (rows r + 8*kh spread across 16-lane halves)
#pragma unroll
        for (int r = 0; r < 8; r++) {
            float tmax = fmaxf(sim[0][r], sim[1][r]);
#pragma unroll
            for (int off = 8; off > 0; off >>= 1)
                tmax = fmaxf(tmax, __shfl_xor(tmax, off, 32));
            float mnew  = fmaxf(mi[r], tmax);
            float alpha = __expf(mi[r] - mnew);
            float p0 = __expf(sim[0][r] - mnew);
            float p1 = __expf(sim[1][r] - mnew);
            float ps = p0 + p1;
#pragma unroll
            for (int off = 8; off > 0; off >>= 1)
                ps += __shfl_xor(ps, off, 32);
            li[r] = li[r] * alpha + ps;
            mi[r] = mnew;
#pragma unroll
            for (int n8 = 0; n8 < 8; n8++) oacc[n8][r] *= alpha;
            Pld[mt][r + 8 * kh][lm]      = (bf16)p0;
            Pld[mt][r + 8 * kh][16 + lm] = (bf16)p1;
        }
        // per-wave LDS ordering: DS ops are in-order within a wave; fence the
        // compiler and drain DScnt before re-reading the freshly written tile.
        asm volatile("s_wait_dscnt 0" ::: "memory");

        // out += P @ V  (A = P 16x32 from LDS, B = vt 32x16 from global)
        v16bf afp;
        {
            const bf16* p = &Pld[mt][lm][kh * 16];
            afp = cat8(*(const v8bf*)p, *(const v8bf*)(p + 8));
        }
#pragma unroll
        for (int n8 = 0; n8 < 8; n8++) {
            const bf16* p = vtb + (size_t)(n8 * 16 + lm) * S_ + s0 + kh * 16;
            oacc[n8] = wmma_bf(afp, ld16(p), oacc[n8]);
        }
        asm volatile("" ::: "memory");   // keep P stores of next iter behind reads
    }

    // write partials
    size_t pbase = ((size_t)(bh * SPLIT_ + sp) * 4 + mt) * (16 * 128);
#pragma unroll
    for (int n8 = 0; n8 < 8; n8++) {
        int d = n8 * 16 + lm;
#pragma unroll
        for (int r = 0; r < 8; r++)
            pout[pbase + (r + 8 * kh) * 128 + d] = oacc[n8][r];
    }
    if (lm == 0) {
        size_t sb = (size_t)(bh * SPLIT_ + sp) * 64 + mt * 16;
#pragma unroll
        for (int r = 0; r < 8; r++) {
            pm[sb + r + 8 * kh] = mi[r];
            pl[sb + r + 8 * kh] = li[r];
        }
    }
}

// ---------------------------------------------------------------------------
// Combine split-softmax partials -> attn output bf16 [B, 64, INNER]
// ---------------------------------------------------------------------------
__global__ __launch_bounds__(128) void attn_combine_kernel(
    const float* __restrict__ pout, const float* __restrict__ pm,
    const float* __restrict__ pl, bf16* __restrict__ attn)
{
    int h = blockIdx.x, i = blockIdx.y, b = blockIdx.z;
    int d = threadIdx.x;
    int bh = b * H_ + h;

    float mx = -3.0e38f;
#pragma unroll
    for (int sp = 0; sp < SPLIT_; sp++)
        mx = fmaxf(mx, pm[(size_t)(bh * SPLIT_ + sp) * 64 + i]);

    float os = 0.f, ls = 0.f;
#pragma unroll
    for (int sp = 0; sp < SPLIT_; sp++) {
        size_t sb = (size_t)(bh * SPLIT_ + sp);
        float w = __expf(pm[sb * 64 + i] - mx);
        ls += w * pl[sb * 64 + i];
        os += w * pout[(sb * 4 + (i >> 4)) * (size_t)(16 * 128) + (size_t)(i & 15) * 128 + d];
    }
    attn[((size_t)b * NLAT + i) * INNER_ + h * DH_ + d] = (bf16)(os / ls);
}

// ---------------------------------------------------------------------------
extern "C" void kernel_launch(void* const* d_in, const int* in_sizes, int n_in,
                              void* d_out, int out_size, void* d_ws, size_t ws_size,
                              hipStream_t stream)
{
    (void)in_sizes; (void)n_in; (void)out_size; (void)ws_size;
    const float* x   = (const float*)d_in[0];
    const float* lat = (const float*)d_in[1];
    const float* gm  = (const float*)d_in[2];
    const float* bm  = (const float*)d_in[3];
    const float* gl  = (const float*)d_in[4];
    const float* bl  = (const float*)d_in[5];
    const float* Wq  = (const float*)d_in[6];
    const float* bq  = (const float*)d_in[7];
    const float* Wkv = (const float*)d_in[8];
    const float* bkv = (const float*)d_in[9];
    const float* Wo  = (const float*)d_in[10];
    const float* bo  = (const float*)d_in[11];
    float* out = (float*)d_out;

    char* ws = (char*)d_ws;
    size_t off = 0;
    auto take = [&](size_t bytes) -> char* {
        char* p = ws + off;
        off = (off + bytes + 255) & ~(size_t)255;
        return p;
    };
    bf16*  kv_in   = (bf16*)take((size_t)B_ * S_ * DIM_ * 2);            // 68 MB
    bf16*  Wq_t    = (bf16*)take((size_t)INNER_ * DIM_ * 2);             // [N][K]
    bf16*  Wkv_t   = (bf16*)take((size_t)KV2_ * DIM_ * 2);               // [N][K]
    bf16*  Wo_t    = (bf16*)take((size_t)DIM_ * INNER_ * 2);             // [N][K]
    bf16*  q_bf    = (bf16*)take((size_t)B_ * NLAT * INNER_ * 2);
    bf16*  k_bf    = (bf16*)take((size_t)B_ * S_ * INNER_ * 2);          // 68 MB
    bf16*  vt_bf   = (bf16*)take((size_t)B_ * H_ * DH_ * S_ * 2);        // 68 MB, [bh*128+d][s]
    bf16*  attn_bf = (bf16*)take((size_t)B_ * NLAT * INNER_ * 2);
    float* pout    = (float*)take((size_t)B_ * H_ * SPLIT_ * 64 * DH_ * 4);
    float* pmax    = (float*)take((size_t)B_ * H_ * SPLIT_ * 64 * 4);
    float* psum    = (float*)take((size_t)B_ * H_ * SPLIT_ * 64 * 4);

    // 1) fused layernorm + concat into bf16 KV-input
    ln_fuse_kernel<<<B_ * S_, 256, 0, stream>>>(x, lat, gm, bm, gl, bl, kv_in);

    // 2) transpose-convert weights to bf16 [N][K]
    cvt_transpose_kernel<<<dim3(INNER_ / 32, DIM_ / 32), 256, 0, stream>>>(Wq,  Wq_t,  DIM_, INNER_);
    cvt_transpose_kernel<<<dim3(KV2_   / 32, DIM_ / 32), 256, 0, stream>>>(Wkv, Wkv_t, DIM_, KV2_);
    cvt_transpose_kernel<<<dim3(DIM_   / 32, INNER_ / 32), 256, 0, stream>>>(Wo,  Wo_t,  INNER_, DIM_);

    // 3) kv projection (139.5 GFLOP): K half row-major, V half transposed
    gemm_wmma_kernel<2><<<dim3(KV2_ / 128, (B_ * S_) / 128), 256, 0, stream>>>(
        kv_in, Wkv_t, bkv, k_bf, vt_bf, DIM_, KV2_, B_ * S_, 0, 0, 1.0f);

    // 4) q = (ln @ Wq + bq) * DH^-0.5 (ln rows live inside kv_in at offset M)
    gemm_wmma_kernel<0><<<dim3(INNER_ / 128, (B_ * NLAT) / 128), 256, 0, stream>>>(
        kv_in, Wq_t, bq, q_bf, nullptr, DIM_, INNER_, NLAT, S_, M_, 0.08838834764831845f);

    // 5) flash attention, split 10x along S
    attn_partial_kernel<<<dim3(SPLIT_, H_, B_), 128, 0, stream>>>(
        q_bf, k_bf, vt_bf, pout, pmax, psum);
    attn_combine_kernel<<<dim3(H_, NLAT, B_), DH_, 0, stream>>>(pout, pmax, psum, attn_bf);

    // 6) out = attn @ Wo + bo  (f32 output)
    gemm_wmma_kernel<1><<<dim3(DIM_ / 128, (B_ * NLAT) / 128), 256, 0, stream>>>(
        attn_bf, Wo_t, bo, out, nullptr, INNER_, DIM_, B_ * NLAT, 0, 0, 1.0f);
}
